// SudokuRRNLatent_29927332119059
// MI455X (gfx1250) — matrix operand order repaired and checked by reading, not compiled, not run on hardware
//
#include <hip/hip_runtime.h>
#include <hip/hip_bf16.h>

// ---------------------------------------------------------------------------
// SudokuRRN latent forward for MI455X (gfx1250, wave32, WMMA).
//
// B=256, N=20736 nodes, E=414720 edges, H=96, STEPS=16, OUT=128.
// Edge list is destination-major (20 contiguous edges per dst node) ->
// deterministic in-block segment-sum, no float atomics.
//
// All GEMMs use v_wmma_f32_16x16x32_f16. All K dims are compile-time
// multiples of 32 (xin zero-padded 48->64) so fragment loads are branch-free
// vectorized b128 loads.
// ---------------------------------------------------------------------------

typedef _Float16 h16;
typedef unsigned int u32;
typedef __attribute__((ext_vector_type(16))) _Float16 v16h;
typedef __attribute__((ext_vector_type(4)))  _Float16 v4h;
typedef __attribute__((ext_vector_type(8)))  float    v8f;

#define BB 256
#define NN (BB * 81)      // 20736 nodes  (divisible by 16)
#define EE (NN * 20)      // 414720 edges (divisible by 16)
#define HH 96

// ---------------------------------------------------------------------------
// Fragment layouts (CDNA5 ISA 7.12.2, 16-bit A 16x32 / B 32x16):
//  A: lane<16 -> row=lane,    element e holds K = koff + (e<8 ? e : 16+(e-8)),
//     koff = (lane&16) ? 8 : 0
//  B: col=lane&15, element e holds the same K as A (koff + (e<8 ? e : 16+(e-8)))
//  C/D: N = lane&15; VGPR r -> M = r + (lane&16 ? 8 : 0)
// Per lane the A elements are two contiguous 8-value K runs -> 4x float4.
// ---------------------------------------------------------------------------

__device__ __forceinline__ v16h load_a_frag_f32(const float* __restrict__ arow,
                                                int kc, int koff) {
    const float4* p0 = (const float4*)(arow + kc + koff);       // k = kc+koff..+7
    const float4* p1 = (const float4*)(arow + kc + 16 + koff);  // k = +16..+23
    float4 f0 = p0[0], f1 = p0[1], f2 = p1[0], f3 = p1[1];
    v16h a;
    a[0]=(h16)f0.x;  a[1]=(h16)f0.y;  a[2]=(h16)f0.z;  a[3]=(h16)f0.w;
    a[4]=(h16)f1.x;  a[5]=(h16)f1.y;  a[6]=(h16)f1.z;  a[7]=(h16)f1.w;
    a[8]=(h16)f2.x;  a[9]=(h16)f2.y;  a[10]=(h16)f2.z; a[11]=(h16)f2.w;
    a[12]=(h16)f3.x; a[13]=(h16)f3.y; a[14]=(h16)f3.z; a[15]=(h16)f3.w;
    return a;
}

// A-fragment from f16 LDS row (halves packed as u32 pairs): two uint4 reads.
__device__ __forceinline__ v16h load_a_frag_lds(const u32* arow, int kc, int koff) {
    const uint4* p0 = (const uint4*)(arow + ((kc + koff) >> 1));
    const uint4* p1 = (const uint4*)(arow + ((kc + 16 + koff) >> 1));
    uint4 u0 = p0[0], u1 = p1[0];
    v16h a;
    u32* au = (u32*)&a;
    au[0] = u0.x; au[1] = u0.y; au[2] = u0.z; au[3] = u0.w;
    au[4] = u1.x; au[5] = u1.y; au[6] = u1.z; au[7] = u1.w;
    return a;
}

// ---------------------------------------------------------------------------
// Pack W[out=Nc][in=K] (row-major f32) into B-fragment-ordered f16:
// Bp[(chunk*ntiles + nt)*512 + lane*16 + e], zero-padded past K.
// ---------------------------------------------------------------------------
__global__ __launch_bounds__(256) void k_pack(const float* __restrict__ W,
                                              h16* __restrict__ Bp,
                                              int K, int Nc) {
    int nch = (K + 31) >> 5, ntl = Nc >> 4;
    int tot = nch * ntl * 512;
    int idx = blockIdx.x * 256 + threadIdx.x;
    if (idx >= tot) return;
    int lane_e = idx & 511;
    int tile   = idx >> 9;
    int ch = tile / ntl, nt = tile % ntl;
    int lane = lane_e >> 4, e = lane_e & 15;
    int k = (ch << 5) + ((lane & 16) ? 8 : 0) + (e < 8 ? e : 8 + e);
    int n = nt * 16 + (lane & 15);
    Bp[idx] = (k < K) ? (h16)W[(size_t)n * K + k] : (h16)0.0f;
}

// Stacked LSTM weight [wih.T ; whh.T] -> [K=288][Nc=384] fragment order.
__global__ __launch_bounds__(256) void k_pack_lstm(const float* __restrict__ wih,
                                                   const float* __restrict__ whh,
                                                   h16* __restrict__ Bp) {
    const int ntl = 24;
    int idx = blockIdx.x * 256 + threadIdx.x;
    if (idx >= 9 * ntl * 512) return;
    int lane_e = idx & 511;
    int tile   = idx >> 9;
    int ch = tile / ntl, nt = tile % ntl;
    int lane = lane_e >> 4, e = lane_e & 15;
    int k = (ch << 5) + ((lane & 16) ? 8 : 0) + (e < 8 ? e : 8 + e);
    int n = nt * 16 + (lane & 15);
    float v = (k < 192) ? wih[(size_t)n * 192 + k]
                        : whh[(size_t)n * 96 + (k - 192)];
    Bp[idx] = (h16)v;
}

// ---------------------------------------------------------------------------
// Input embedding, written PADDED to 64 columns (cols 48..63 = 0) so the
// first GEMM needs no K bounds checks.
// ---------------------------------------------------------------------------
__global__ __launch_bounds__(256) void k_xin(const float* __restrict__ yhat,
                                             const int* __restrict__ target,
                                             const float* __restrict__ de,
                                             const float* __restrict__ re,
                                             const float* __restrict__ ce,
                                             float* __restrict__ xin) {
    int idx = blockIdx.x * 256 + threadIdx.x;
    if (idx >= NN * 64) return;
    int n = idx >> 6, col = idx & 63;
    int b = n / 81, cell = n % 81;
    float v = 0.0f;
    if (col < 16) {
        const float* y = yhat + (size_t)b * 810 + cell;   // [B,10,81], stride 81
        float s = 0.0f;
        #pragma unroll
        for (int d = 0; d < 10; ++d) s += y[d * 81] * de[d * 16 + col];
        v = s - de[target[n] * 16 + col];
    } else if (col < 32) {
        v = re[(cell / 9) * 16 + (col - 16)];
    } else if (col < 48) {
        v = ce[(cell % 9) * 16 + (col - 32)];
    }
    xin[idx] = v;
}

// ---------------------------------------------------------------------------
// Wave-tile WMMA GEMM, compile-time K/Nc: C[M][Nc] = act(A[M][K] @ Bp + bias)
// ---------------------------------------------------------------------------
template <int K, int NC>
__global__ __launch_bounds__(128) void k_gemm(const float* __restrict__ A,
                                              const h16* __restrict__ Bp,
                                              const float* __restrict__ bias,
                                              float* __restrict__ C,
                                              int M, int relu) {
    const int ntiles = NC / 16;
    const int nch    = K / 32;
    int wave = (blockIdx.x * 128 + threadIdx.x) >> 5;
    int lane = threadIdx.x & 31;
    if (wave >= (M >> 4) * ntiles) return;        // wave-uniform: EXEC all-1 inside
    int mt = wave / ntiles, nt = wave % ntiles;
    int row  = mt * 16 + (lane & 15);
    int koff = (lane & 16) ? 8 : 0;
    const float* arow = A + (size_t)row * K;
    v8f acc = {};
    #pragma unroll
    for (int ch = 0; ch < nch; ++ch) {
        v16h a = load_a_frag_f32(arow, ch * 32, koff);
        v16h b = *(const v16h*)(Bp + ((size_t)(ch * ntiles + nt)) * 512 + lane * 16);
        acc = __builtin_amdgcn_wmma_f32_16x16x32_f16(false, a, false, b,
                                                     (short)0, acc, false, false);
    }
    int n  = nt * 16 + (lane & 15);
    float bv = bias ? bias[n] : 0.0f;
    int mb = mt * 16 + ((lane & 16) ? 8 : 0);
    #pragma unroll
    for (int r = 0; r < 8; ++r) {
        float v = acc[r] + bv;
        if (relu) v = fmaxf(v, 0.0f);
        C[(size_t)(mb + r) * NC + n] = v;
    }
}

// LSTM gates GEMM: A rows virtually concat [x | m | rh] (K=288, Nc=384).
// Each 32-wide K chunk lies wholly inside one 96-col source buffer.
__global__ __launch_bounds__(128) void k_gemm_lstm(const float* __restrict__ X,
                                                   const float* __restrict__ Mm,
                                                   const float* __restrict__ Rh,
                                                   const h16* __restrict__ Bp,
                                                   float* __restrict__ C) {
    const int ntiles = 24;
    int wave = (blockIdx.x * 128 + threadIdx.x) >> 5;
    int lane = threadIdx.x & 31;
    if (wave >= (NN >> 4) * ntiles) return;
    int mt = wave / ntiles, nt = wave % ntiles;
    int row  = mt * 16 + (lane & 15);
    int koff = (lane & 16) ? 8 : 0;
    const float* rowX = X  + (size_t)row * HH;
    const float* rowM = Mm + (size_t)row * HH;
    const float* rowR = Rh + (size_t)row * HH;
    v8f acc = {};
    #pragma unroll
    for (int ch = 0; ch < 9; ++ch) {
        const float* arow = (ch < 3) ? rowX : (ch < 6) ? rowM : rowR;
        int kc = (ch % 3) * 32;
        v16h a = load_a_frag_f32(arow, kc, koff);
        v16h b = *(const v16h*)(Bp + ((size_t)(ch * ntiles + nt)) * 512 + lane * 16);
        acc = __builtin_amdgcn_wmma_f32_16x16x32_f16(false, a, false, b,
                                                     (short)0, acc, false, false);
    }
    int n  = nt * 16 + (lane & 15);
    int mb = mt * 16 + ((lane & 16) ? 8 : 0);
    #pragma unroll
    for (int r = 0; r < 8; ++r)
        C[(size_t)(mb + r) * 384 + n] = acc[r];
}

// ---------------------------------------------------------------------------
// Fused edge-message MLP + segment-sum.
// Block = 80 edges = 4 dst nodes x 20 edges; 4 waves x 30 (5Mx6N) tile-tasks
// per layer; activations ping-pong in LDS; final f32 tile + in-block reduce.
// ---------------------------------------------------------------------------
__global__ __launch_bounds__(128) void k_msg(const float* __restrict__ H,
                                             const int* __restrict__ src,
                                             const int* __restrict__ dst,
                                             const h16* __restrict__ w0,
                                             const h16* __restrict__ w1,
                                             const h16* __restrict__ w2,
                                             const h16* __restrict__ w3,
                                             const float* __restrict__ b0,
                                             const float* __restrict__ b1,
                                             const float* __restrict__ b2,
                                             const float* __restrict__ b3,
                                             float* __restrict__ Mout) {
    __shared__ char smem[80 * 192 * 2 + 80 * 96 * 2];      // 46080 B
    h16*   actA = (h16*)smem;                              // [80][192] f16
    h16*   actB = (h16*)(smem + 80 * 192 * 2);             // [80][96]  f16
    float* msgf = (float*)smem;                            // [80][96]  f32 (aliases actA)

    const h16*   Ws[4] = {w0, w1, w2, w3};
    const float* Bs[4] = {b0, b1, b2, b3};

    int e0  = blockIdx.x * 80;
    int tid = threadIdx.x;

    // Stage layer-0 input: [h[src] | h[dst]] per edge. float4 gather loads,
    // packed v4h LDS stores. Column math: act col = c4*4 for both halves.
    for (int idx = tid; idx < 80 * 48; idx += 128) {
        int r = idx / 48, c4 = idx - r * 48;               // c4 in [0,48): 4-col groups
        int e = e0 + r;
        int node = (c4 < 24) ? src[e] : dst[e];
        int c = (c4 < 24 ? c4 : c4 - 24) * 4;
        float4 f = *(const float4*)(H + (size_t)node * HH + c);
        v4h h4 = {(h16)f.x, (h16)f.y, (h16)f.z, (h16)f.w};
        *(v4h*)(actA + r * 192 + c4 * 4) = h4;
    }
    __syncthreads();

    int wv = tid >> 5, lane = tid & 31;
    #pragma unroll
    for (int l = 0; l < 4; ++l) {
        const int Kl  = (l == 0) ? 192 : 96;
        const int nch = Kl / 32;
        const h16* in  = ((l & 1) == 0) ? actA : actB;
        h16*       out = ((l & 1) == 0) ? actB : actA;
        for (int task = wv; task < 30; task += 4) {        // 5 Mtiles x 6 Ntiles
            int mt = task / 6, nt = task - mt * 6;
            int rrow = mt * 16 + (lane & 15);
            int koff = (lane & 16) ? 8 : 0;
            const u32* arow = (const u32*)(in + (size_t)rrow * Kl);
            v8f acc = {};
            #pragma unroll
            for (int ch = 0; ch < nch; ++ch) {
                v16h a = load_a_frag_lds(arow, ch * 32, koff);
                v16h b = *(const v16h*)(Ws[l] + ((size_t)(ch * 6 + nt)) * 512 + lane * 16);
                acc = __builtin_amdgcn_wmma_f32_16x16x32_f16(false, a, false, b,
                                                             (short)0, acc, false, false);
            }
            int n  = nt * 16 + (lane & 15);
            float bv = Bs[l][n];
            int mb = mt * 16 + ((lane & 16) ? 8 : 0);
            if (l < 3) {
                #pragma unroll
                for (int r = 0; r < 8; ++r)
                    out[(size_t)(mb + r) * 96 + n] = (h16)fmaxf(acc[r] + bv, 0.0f);
            } else {                                       // final layer -> f32
                #pragma unroll
                for (int r = 0; r < 8; ++r)
                    msgf[(size_t)(mb + r) * 96 + n] = acc[r] + bv;
            }
        }
        __syncthreads();
    }

    // Deterministic segment-sum: 4 nodes x 96 cols, 20 contiguous edges each.
    for (int idx = tid; idx < 4 * 96; idx += 128) {
        int q = idx / 96, c = idx - q * 96;
        float s = 0.0f;
        #pragma unroll
        for (int j = 0; j < 20; ++j) s += msgf[(size_t)(q * 20 + j) * 96 + c];
        int node = dst[e0 + q * 20];                       // == blockIdx*4 + q
        Mout[(size_t)node * HH + c] = s;
    }
}

// ---------------------------------------------------------------------------
// LSTM cell elementwise update (torch gate order i,f,g,o; no bias).
// ---------------------------------------------------------------------------
__global__ __launch_bounds__(256) void k_lstm(const float* __restrict__ gates,
                                              float* __restrict__ cst,
                                              float* __restrict__ hbuf,
                                              float* __restrict__ rbuf) {
    int idx = blockIdx.x * 256 + threadIdx.x;
    if (idx >= NN * HH) return;
    int n = idx / HH, c = idx - n * HH;
    const float* g = gates + (size_t)n * 384;
    float gi = g[c], gf = g[96 + c], gg = g[192 + c], go = g[288 + c];
    float si = 1.0f / (1.0f + expf(-gi));
    float sf = 1.0f / (1.0f + expf(-gf));
    float so = 1.0f / (1.0f + expf(-go));
    float nc = sf * cst[idx] + si * tanhf(gg);
    float nh = so * tanhf(nc);
    cst[idx]  = nc;
    hbuf[idx] = nh;
    rbuf[idx] = nh;
}

// ---------------------------------------------------------------------------
// Readout: per-batch max-pool, self-attention softmax, pooled projection.
// ---------------------------------------------------------------------------
__global__ __launch_bounds__(128) void k_readout(const float* __restrict__ H,
                                                 const float* __restrict__ ow,
                                                 const float* __restrict__ ob,
                                                 float* __restrict__ out) {
    __shared__ float hr[81 * 96];
    __shared__ float mp[96];
    __shared__ float sc[81];
    __shared__ float ssum[1];
    int b = blockIdx.x, tid = threadIdx.x;
    for (int idx = tid; idx < 81 * 96; idx += 128)
        hr[idx] = H[(size_t)(b * 81) * HH + idx];
    __syncthreads();
    if (tid < 96) {
        float m = -3.4e38f;
        for (int j = 0; j < 81; ++j) m = fmaxf(m, hr[j * 96 + tid]);
        mp[tid] = m;
    }
    __syncthreads();
    if (tid < 81) {
        float s = 0.0f;
        for (int c = 0; c < 96; ++c) s += hr[tid * 96 + c] * mp[c];
        sc[tid] = s * (1.0f / 96.0f);
    }
    __syncthreads();
    if (tid == 0) {
        float mx = -3.4e38f;
        for (int j = 0; j < 81; ++j) mx = fmaxf(mx, sc[j]);
        float sum = 0.0f;
        for (int j = 0; j < 81; ++j) { float e = expf(sc[j] - mx); sc[j] = e; sum += e; }
        ssum[0] = sum;
    }
    __syncthreads();
    if (tid < 96) {
        float inv = 1.0f / ssum[0];
        float s = 0.0f;
        for (int j = 0; j < 81; ++j) s += sc[j] * inv * hr[j * 96 + tid];
        mp[tid] = s;                                       // pooled
    }
    __syncthreads();
    if (tid < 128) {
        float s = 0.0f;
        for (int c = 0; c < 96; ++c) s += ow[tid * 96 + c] * mp[c];
        out[(size_t)b * 128 + tid] = s + ob[tid];
    }
}

// ---------------------------------------------------------------------------
// Host launcher
// ---------------------------------------------------------------------------
extern "C" void kernel_launch(void* const* d_in, const int* in_sizes, int n_in,
                              void* d_out, int out_size, void* d_ws, size_t ws_size,
                              hipStream_t stream) {
    (void)in_sizes; (void)n_in; (void)out_size; (void)ws_size;

    const float* y_hat  = (const float*)d_in[0];
    const int*   target = (const int*)d_in[1];
    const int*   src    = (const int*)d_in[2];
    const int*   dst    = (const int*)d_in[3];
    const float* de     = (const float*)d_in[4];
    const float* re     = (const float*)d_in[5];
    const float* ce     = (const float*)d_in[6];
    const float* in_w[4]  = {(const float*)d_in[7],  (const float*)d_in[9],
                             (const float*)d_in[11], (const float*)d_in[13]};
    const float* in_b[4]  = {(const float*)d_in[8],  (const float*)d_in[10],
                             (const float*)d_in[12], (const float*)d_in[14]};
    const float* msg_w[4] = {(const float*)d_in[15], (const float*)d_in[17],
                             (const float*)d_in[19], (const float*)d_in[21]};
    const float* msg_b[4] = {(const float*)d_in[16], (const float*)d_in[18],
                             (const float*)d_in[20], (const float*)d_in[22]};
    const float* wih   = (const float*)d_in[23];
    const float* whh   = (const float*)d_in[24];
    const float* out_w = (const float*)d_in[25];
    const float* out_b = (const float*)d_in[26];
    float* outp = (float*)d_out;

    // ---- workspace carve-up (256B aligned) ----
    char* ws = (char*)d_ws;
    size_t off = 0;
    auto take = [&](size_t bytes) -> char* {
        char* p = ws + off;
        off += (bytes + 255) & ~(size_t)255;
        return p;
    };
    float* xin   = (float*)take((size_t)NN * 64 * 4);   // padded 48->64
    float* x     = (float*)take((size_t)NN * HH * 4);
    float* hbuf  = (float*)take((size_t)NN * HH * 4);
    float* rbuf  = (float*)take((size_t)NN * HH * 4);
    float* cbuf  = (float*)take((size_t)NN * HH * 4);
    float* mbuf  = (float*)take((size_t)NN * HH * 4);
    float* gates = (float*)take((size_t)NN * 384 * 4);
    float* t0    = (float*)take((size_t)NN * HH * 4);
    float* t1    = (float*)take((size_t)NN * HH * 4);
    h16* pw_in[4];
    pw_in[0] = (h16*)take((size_t)2 * 6 * 512 * 2);
    for (int k = 1; k < 4; ++k) pw_in[k] = (h16*)take((size_t)3 * 6 * 512 * 2);
    h16* pw_msg[4];
    pw_msg[0] = (h16*)take((size_t)6 * 6 * 512 * 2);
    for (int k = 1; k < 4; ++k) pw_msg[k] = (h16*)take((size_t)3 * 6 * 512 * 2);
    h16* pw_lstm = (h16*)take((size_t)9 * 24 * 512 * 2);

    // ---- weight packing ----
    k_pack<<<(2 * 6 * 512 + 255) / 256, 256, 0, stream>>>(in_w[0], pw_in[0], 48, 96);
    for (int k = 1; k < 4; ++k)
        k_pack<<<(3 * 6 * 512 + 255) / 256, 256, 0, stream>>>(in_w[k], pw_in[k], 96, 96);
    k_pack<<<(6 * 6 * 512 + 255) / 256, 256, 0, stream>>>(msg_w[0], pw_msg[0], 192, 96);
    for (int k = 1; k < 4; ++k)
        k_pack<<<(3 * 6 * 512 + 255) / 256, 256, 0, stream>>>(msg_w[k], pw_msg[k], 96, 96);
    k_pack_lstm<<<(9 * 24 * 512 + 255) / 256, 256, 0, stream>>>(wih, whh, pw_lstm);

    // ---- input embedding + input MLP ----
    k_xin<<<(NN * 64 + 255) / 256, 256, 0, stream>>>(y_hat, target, de, re, ce, xin);
    int blk96 = (((NN >> 4) * 6) + 3) / 4;              // 1944 blocks for Nc=96
    k_gemm<64, 96><<<blk96, 128, 0, stream>>>(xin, pw_in[0], in_b[0], t0, NN, 1);
    k_gemm<96, 96><<<blk96, 128, 0, stream>>>(t0,  pw_in[1], in_b[1], t1, NN, 1);
    k_gemm<96, 96><<<blk96, 128, 0, stream>>>(t1,  pw_in[2], in_b[2], t0, NN, 1);
    k_gemm<96, 96><<<blk96, 128, 0, stream>>>(t0,  pw_in[3], in_b[3], x,  NN, 0);

    // carry init: hg = x, rh = 0, c = 0
    hipMemcpyAsync(hbuf, x, (size_t)NN * HH * 4, hipMemcpyDeviceToDevice, stream);
    hipMemsetAsync(rbuf, 0, (size_t)NN * HH * 4, stream);
    hipMemsetAsync(cbuf, 0, (size_t)NN * HH * 4, stream);

    // ---- 16 recurrent steps ----
    int lstm_blocks = (((NN >> 4) * 24) + 3) / 4;        // 7776
    for (int s = 0; s < 16; ++s) {
        k_msg<<<NN / 4, 128, 0, stream>>>(hbuf, src, dst,
                                          pw_msg[0], pw_msg[1], pw_msg[2], pw_msg[3],
                                          msg_b[0], msg_b[1], msg_b[2], msg_b[3],
                                          mbuf);
        k_gemm_lstm<<<lstm_blocks, 128, 0, stream>>>(x, mbuf, rbuf, pw_lstm, gates);
        k_lstm<<<(NN * HH + 255) / 256, 256, 0, stream>>>(gates, cbuf, hbuf, rbuf);
    }

    // ---- attention readout ----
    k_readout<<<BB, 128, 0, stream>>>(hbuf, out_w, out_b, outp);
}
